// sparse_resblock_76811195121951
// MI455X (gfx1250) — matrix operand order repaired and checked by reading, not compiled, not run on hardware
//
#include <hip/hip_runtime.h>

typedef __bf16 bf16;
typedef bf16  v16bf __attribute__((ext_vector_type(16)));
typedef bf16  v8bf  __attribute__((ext_vector_type(8)));
typedef float v8f   __attribute__((ext_vector_type(8)));
typedef unsigned int v4u __attribute__((ext_vector_type(4)));
typedef int  v8i __attribute__((ext_vector_type(8)));
typedef int  v4i __attribute__((ext_vector_type(4)));

constexpr int   NPTS = 131072;
constexpr int   C    = 128;
constexpr int   KOFF = 27;
constexpr int   MMAP = 65536;
constexpr float EPS  = 1e-5f;

constexpr int TPB_TILES      = 8;                 // 128-row tiles per block
constexpr int ROWS_PER_BLOCK = 128 * TPB_TILES;   // 1024 map rows / block

// Non-returning f32 atomic add (global_atomic_add_f32, no CAS fallback).
__device__ __forceinline__ void atomic_add_f32(float* p, float v) {
  asm volatile("global_atomic_add_f32 %0, %1, off" :: "v"(p), "v"(v) : "memory");
}

// ---------------------------------------------------------------------------
// TDM: DMA one pre-swizzled 32KB bf16 weight tile from global into LDS.
// D# per CDNA5 ISA ch.8: group0 = {count/flags, lds_addr, global_addr, type},
// group1 = {mask/data_size, tensor_dim0, tensor_dim1, tile_dim0, tile_dim1,
// tensor_dim0_stride, ...}. 1-D copy: data_size=8B, dim0 = 4096 elements.
// ---------------------------------------------------------------------------
__device__ __forceinline__ void tdm_load_w_tile(const bf16* src, bf16* dst_lds) {
#if __has_builtin(__builtin_amdgcn_tensor_load_to_lds)
  const unsigned long long ga = (unsigned long long)(uintptr_t)src;
  const unsigned lds_off = (unsigned)(uintptr_t)dst_lds;   // low 32 bits = LDS byte offset
  const v4u g0 = {
      1u,                                                  // count=1 (valid), user mode
      lds_off,                                             // bits[63:32]  lds_addr
      (unsigned)(ga & 0xffffffffu),                        // bits[95:64]  global_addr[31:0]
      (unsigned)((ga >> 32) & 0x01ffffffu) | (2u << 30)    // global_addr[56:32] | type=2
  };
  const v8i g1 = {
      0x30000,               // workgroup_mask=0 | data_size=3 (8 bytes)
      (int)(4096u << 16),    // tensor_dim0 = 4096 (8B units)
      1 << 16,               // tensor_dim1 = 1
      (int)(4096u << 16),    // tile_dim0   = 4096
      1,                     // tile_dim1   = 1, tile_dim2 = 0
      4096,                  // tensor_dim0_stride = 4096
      0, 0
  };
  const v4i gz = {0, 0, 0, 0};
#if __clang_major__ >= 23
  const v8i gz8 = {0, 0, 0, 0, 0, 0, 0, 0};
  __builtin_amdgcn_tensor_load_to_lds(g0, g1, gz, gz, gz8, 0);
#else
  __builtin_amdgcn_tensor_load_to_lds(g0, g1, gz, gz, 0);
#endif
  __builtin_amdgcn_s_wait_tensorcnt(0);
#else
  // Fallback: cooperative wave copy (one wave executes this).
  for (int q = (int)(threadIdx.x & 31); q < (C * C) / 8; q += 32)
    *(v8bf*)&dst_lds[q * 8] = *(const v8bf*)&src[q * 8];
#endif
}

// ---------------------------------------------------------------------------
// One-time weight prep: W[k] (f32 row-major, [c_in][c_out]) -> bf16,
// transposed to Wt[n][c_in] with 8-element chunks XOR-swizzled by (n & 15).
// This is byte-for-byte the LDS image conv_wmma consumes, so the TDM copy
// is a straight linear DMA. grid = 2*KOFF blocks (W1 then W2).
// ---------------------------------------------------------------------------
__global__ __launch_bounds__(256) void prep_w(
    const float* __restrict__ W1, const float* __restrict__ W2,
    bf16* __restrict__ Wt)
{
  const int b = blockIdx.x;
  const float* Wk = (b < KOFF) ? (W1 + (size_t)b * C * C)
                               : (W2 + (size_t)(b - KOFF) * C * C);
  bf16* dst = Wt + (size_t)b * C * C;
  for (int q = threadIdx.x; q < (C * C) / 8; q += blockDim.x) {
    const int n  = q >> 4;    // output channel
    const int ch = q & 15;    // 8-element chunk along c_in
    v8bf pk;
#pragma unroll
    for (int j = 0; j < 8; ++j)
      pk[j] = (bf16)Wk[(ch * 8 + j) * C + n];
    *(v8bf*)&dst[n * C + ((ch ^ (n & 15)) * 8)] = pk;
  }
}

// ---------------------------------------------------------------------------
// Gather -> (M x 128) @ W[k] (128x128) -> scatter-add, bf16 WMMA, f32 accum.
// grid.x = KOFF * (MMAP / ROWS_PER_BLOCK), block = 256 threads (8 waves).
// Each wave computes a 16-row x 128-col output stripe per tile.
// Wt = pre-swizzled bf16 weights for this conv's 27 offsets.
// ---------------------------------------------------------------------------
__global__ __launch_bounds__(256) void conv_wmma(
    const float* __restrict__ x, const bf16* __restrict__ Wt,
    const int* __restrict__ map_in, const int* __restrict__ map_out,
    float* __restrict__ out)
{
  __shared__ bf16 sW[C * C];   // swizzled bf16 W[k] tile (32 KB)

  const int blocks_per_k = MMAP / ROWS_PER_BLOCK;
  const int k   = blockIdx.x / blocks_per_k;
  const int blk = blockIdx.x % blocks_per_k;
  const int*  min_k  = map_in  + (size_t)k * MMAP;
  const int*  mout_k = map_out + (size_t)k * MMAP;

  // Wave 0 DMAs the weight tile via the Tensor Data Mover.
  if (threadIdx.x < 32)
    tdm_load_w_tile(Wt + (size_t)k * C * C, sW);
  __syncthreads();

  const int lane = threadIdx.x & 31;
  const int wv   = threadIdx.x >> 5;
  const int nl   = lane & 15;
  const int hi   = lane >> 4;          // which half-wave
  const int tile_base = blk * ROWS_PER_BLOCK;

  for (int t = 0; t < TPB_TILES; ++t) {
    const int rbase  = tile_base + t * 128 + wv * 16;
    const int in_idx = min_k[rbase + nl];
    const float* xr  = x + (size_t)in_idx * C;

    v8f acc[8] = {};

#pragma unroll
    for (int kb = 0; kb < 4; ++kb) {
      // ---- A fragment (16x32 bf16), ISA layout:
      // lanes 0-15: K = kb*32 + {0..7, 16..23}; lanes 16-31: {8..15, 24..31}
      const int c0 = kb * 32 + hi * 8;
      const float4 p0 = *(const float4*)(xr + c0);
      const float4 p1 = *(const float4*)(xr + c0 + 4);
      const float4 p2 = *(const float4*)(xr + c0 + 16);
      const float4 p3 = *(const float4*)(xr + c0 + 20);
      v16bf a;
      a[0]=(bf16)p0.x;  a[1]=(bf16)p0.y;  a[2]=(bf16)p0.z;  a[3]=(bf16)p0.w;
      a[4]=(bf16)p1.x;  a[5]=(bf16)p1.y;  a[6]=(bf16)p1.z;  a[7]=(bf16)p1.w;
      a[8]=(bf16)p2.x;  a[9]=(bf16)p2.y;  a[10]=(bf16)p2.z; a[11]=(bf16)p2.w;
      a[12]=(bf16)p3.x; a[13]=(bf16)p3.y; a[14]=(bf16)p3.z; a[15]=(bf16)p3.w;

#pragma unroll
      for (int nt = 0; nt < 8; ++nt) {
        // ---- B fragment (32x16 bf16), ISA layout:
        // lane n (0-15): col N=n, K = kb*32 + {0..7, 16..23}; lanes 16-31: {8..15, 24..31}
        const int row    = nt * 16 + nl;
        const int cbase  = kb * 4 + hi;
        const int chunk0 = (cbase)     ^ nl;
        const int chunk1 = (cbase + 2) ^ nl;
        const v8bf b0 = *(const v8bf*)&sW[row * C + chunk0 * 8];
        const v8bf b1 = *(const v8bf*)&sW[row * C + chunk1 * 8];
        const v16bf b = __builtin_shufflevector(
            b0, b1, 0,1,2,3,4,5,6,7,8,9,10,11,12,13,14,15);
        acc[nt] = __builtin_amdgcn_wmma_f32_16x16x32_bf16(
            false, a, false, b, (short)0, acc[nt], false, false);
      }
    }

    // ---- scatter-add: C/D layout -> VGPR j holds (M=j, lanes 0-15) and
    // (M=j+8, lanes 16-31), N = lane & 15 within each 16-col tile.
    const int* mo = mout_k + rbase;
#pragma unroll
    for (int j = 0; j < 8; ++j) {
      const int row  = j + hi * 8;
      const int oidx = mo[row];
      float* orow = out + (size_t)oidx * C + nl;
#pragma unroll
      for (int nt = 0; nt < 8; ++nt)
        atomic_add_f32(orow + nt * 16, acc[nt][j]);
    }
  }
}

// ---------------------------------------------------------------------------
// Per-channel sum / sum-of-squares. grid = 512, block = 256 (2 rows x 128 ch).
// ---------------------------------------------------------------------------
__global__ __launch_bounds__(256) void bn_stats(
    const float* __restrict__ y, float* __restrict__ ssum, float* __restrict__ ssq)
{
  __shared__ float ls[256], lq[256];
  const int c    = threadIdx.x & (C - 1);
  const int half = threadIdx.x >> 7;
  const int rows = NPTS / gridDim.x;
  const int r0   = blockIdx.x * rows;
  float s = 0.f, q = 0.f;
  for (int r = r0 + half; r < r0 + rows; r += 2) {
    const float v = y[(size_t)r * C + c];
    s += v; q += v * v;
  }
  ls[threadIdx.x] = s; lq[threadIdx.x] = q;
  __syncthreads();
  if (half == 0) {
    s += ls[threadIdx.x + 128];
    q += lq[threadIdx.x + 128];
    atomic_add_f32(&ssum[c], s);
    atomic_add_f32(&ssq[c], q);
  }
}

// ---------------------------------------------------------------------------
// In-place BN + optional residual + ELU over N*C elements.
// ---------------------------------------------------------------------------
__global__ __launch_bounds__(256) void bn_apply(
    float* __restrict__ y, const float* __restrict__ ssum,
    const float* __restrict__ ssq, const float* __restrict__ gamma,
    const float* __restrict__ beta, const float* __restrict__ residual)
{
  const size_t i = (size_t)blockIdx.x * blockDim.x + threadIdx.x;
  const int    c = (int)(i & (C - 1));
  const float inv_n = 1.0f / (float)NPTS;
  const float mean  = ssum[c] * inv_n;
  const float var   = ssq[c] * inv_n - mean * mean;
  const float sc    = rsqrtf(var + EPS) * gamma[c];
  const float sh    = beta[c] - mean * sc;
  float v = y[i] * sc + sh;
  if (residual) v += residual[i];
  y[i] = (v > 0.f) ? v : expm1f(v);
}

// ---------------------------------------------------------------------------
extern "C" void kernel_launch(void* const* d_in, const int* in_sizes, int n_in,
                              void* d_out, int out_size, void* d_ws, size_t ws_size,
                              hipStream_t stream) {
  const float* x      = (const float*)d_in[0];
  const float* W1     = (const float*)d_in[1];
  const float* gamma1 = (const float*)d_in[2];
  const float* beta1  = (const float*)d_in[3];
  const float* W2     = (const float*)d_in[4];
  const float* gamma2 = (const float*)d_in[5];
  const float* beta2  = (const float*)d_in[6];
  const int* map1_in  = (const int*)d_in[7];
  const int* map1_out = (const int*)d_in[8];
  const int* map2_in  = (const int*)d_in[9];
  const int* map2_out = (const int*)d_in[10];
  float* out = (float*)d_out;

  const size_t feat_bytes = (size_t)NPTS * C * sizeof(float);
  float* y1    = (float*)d_ws;                               // N*C f32
  float* stats = (float*)((char*)d_ws + feat_bytes);         // 4*C f32
  bf16*  wt    = (bf16*)((char*)d_ws + feat_bytes + 4096);   // 2*27*C*C bf16
  float* sum1 = stats;
  float* sq1  = stats + C;
  float* sum2 = stats + 2 * C;
  float* sq2  = stats + 3 * C;
  bf16* wt1 = wt;
  bf16* wt2 = wt + (size_t)KOFF * C * C;

  hipMemsetAsync(y1,    0, feat_bytes, stream);
  hipMemsetAsync(stats, 0, 4 * C * sizeof(float), stream);
  hipMemsetAsync(out,   0, feat_bytes, stream);

  const int conv_grid = KOFF * (MMAP / ROWS_PER_BLOCK);   // 27 * 64
  const int ew_grid   = (NPTS * C) / 256;                 // 65536

  // one-time bf16 transpose/swizzle of both weight stacks
  hipLaunchKernelGGL(prep_w, dim3(2 * KOFF), dim3(256), 0, stream, W1, W2, wt);

  // conv1 -> y1
  hipLaunchKernelGGL(conv_wmma, dim3(conv_grid), dim3(256), 0, stream,
                     x, wt1, map1_in, map1_out, y1);
  // BN1 + ELU (in place)
  hipLaunchKernelGGL(bn_stats, dim3(512), dim3(256), 0, stream, y1, sum1, sq1);
  hipLaunchKernelGGL(bn_apply, dim3(ew_grid), dim3(256), 0, stream,
                     y1, sum1, sq1, gamma1, beta1, (const float*)nullptr);
  // conv2 -> out
  hipLaunchKernelGGL(conv_wmma, dim3(conv_grid), dim3(256), 0, stream,
                     y1, wt2, map2_in, map2_out, out);
  // BN2 + residual + ELU (in place)
  hipLaunchKernelGGL(bn_stats, dim3(512), dim3(256), 0, stream, out, sum2, sq2);
  hipLaunchKernelGGL(bn_apply, dim3(ew_grid), dim3(256), 0, stream,
                     out, sum2, sq2, gamma2, beta2, x);
}
